// MFNL_module_40389872451626
// MI455X (gfx1250) — matrix operand order repaired and checked by reading, not compiled, not run on hardware
//
#include <hip/hip_runtime.h>
#include <hip/hip_bf16.h>
#include <stdint.h>

typedef __attribute__((ext_vector_type(16))) __bf16 bf16x16;
typedef __attribute__((ext_vector_type(8)))  __bf16 bf16x8;
typedef __attribute__((ext_vector_type(2)))  __bf16 bf16x2;
typedef __attribute__((ext_vector_type(8)))  float  f32x8;
typedef __attribute__((ext_vector_type(4)))  float  f32x4;
typedef __attribute__((ext_vector_type(4)))  int    i32x4;

#define WMMA_BF16(A, Bm, Cm) \
  __builtin_amdgcn_wmma_f32_16x16x32_bf16(false, (A), false, (Bm), (short)0, (Cm), false, false)

constexpr int BATCH = 16;
constexpr int C     = 512;
constexpr int CR    = 256;
constexpr int SP    = 4096;   // 64*64 spatial per batch

// K-group swizzle for the 16-bit A-operand layout:
// element e<8 -> K = h*8+e, e>=8 -> K = 16+h*8+(e-8). Storing K-groups of 8 in
// order {0-7, 16-23, 8-15, 24-31} makes each lane's fragment a contiguous 32B read.
__device__ __forceinline__ int swz8(int q) { return ((q & 1) << 4) | ((q >> 1) << 3); }

// ---- async global->LDS 16B copy (ASYNCcnt path), with portable fallback ----
// Probe-verified (round 2/3): builtin exists, 4 args, generic i32x4* pointers.
#if defined(__has_builtin)
#if __has_builtin(__builtin_amdgcn_global_load_async_to_lds_b128)
#define HAVE_ASYNC_LDS 1
#endif
#endif

#if defined(HAVE_ASYNC_LDS)
#define ASYNC_CP_B128(gsrc, ldst)                                              \
  __builtin_amdgcn_global_load_async_to_lds_b128((i32x4*)(gsrc), (i32x4*)(ldst), 0, 0)
#if __has_builtin(__builtin_amdgcn_s_wait_asynccnt)
#define ASYNC_WAIT() __builtin_amdgcn_s_wait_asynccnt(0)
#else
#define ASYNC_WAIT() asm volatile("s_wait_asynccnt 0x0" ::: "memory")
#endif
#else
#define ASYNC_CP_B128(gsrc, ldst)                                              \
  (*(bf16x8*)(ldst) = *(const bf16x8*)(gsrc))
#define ASYNC_WAIT() ((void)0)
#endif

// ---------------------------------------------------------------------------
// Kernel 1: fused Q/K/V 1x1-conv projection.  Out[768, 4096] = Wqkv @ Xb + b
// Block: 256 threads (8 waves); tile 128 rows x 128 cols; K-step 32.
// Ping-pong LDS: stage tile kk+1 while WMMA consumes tile kk (1 barrier/iter).
// ---------------------------------------------------------------------------
__global__ __launch_bounds__(256) void qkv_proj_kernel(
    const float* __restrict__ x,
    const float* __restrict__ Wq, const float* __restrict__ bq,
    const float* __restrict__ Wk, const float* __restrict__ bk,
    const float* __restrict__ Wv, const float* __restrict__ bv,
    __bf16* __restrict__ Qo, __bf16* __restrict__ Ko, __bf16* __restrict__ Vo)
{
  __shared__ __align__(16) __bf16 sB[2][128 * 40];  // [col][K], stride 40 (16B aligned)

  const int tid  = threadIdx.x;
  const int lane = tid & 31;
  const int wave = tid >> 5;
  const int b        = blockIdx.z;
  const int colbase  = blockIdx.x * 128;
  const int row0     = blockIdx.y * 128 + wave * 16;   // 0..752 across Q|K|V
  const int seg      = row0 >> 8;                      // 0=Q 1=K 2=V
  const int lrow0    = row0 & 255;

  const float* W    = (seg == 0) ? Wq : (seg == 1) ? Wk : Wv;
  const float* bias = (seg == 0) ? bq : (seg == 1) ? bk : bv;
  __bf16*      Out  = (seg == 0) ? Qo : (seg == 1) ? Ko : Vo;
  const float* xb   = x + (size_t)b * C * SP;

  const int h    = lane >> 4;
  const int l16  = lane & 15;
  const int arow = lrow0 + l16;

  constexpr int KT = C / 32;

  // stage X tile [32 K x 128 cols] fp32 -> bf16 transposed to [col][K];
  // float4 from two adjacent K-rows -> packed bf16x2 b32 stores.
  auto stage = [&](int kk, int buf) {
#pragma unroll
    for (int i = 0; i < 2; ++i) {
      const int u   = tid + i * 256;   // 0..511
      const int kp  = u >> 5;          // K-row pair
      const int c4  = u & 31;          // float4 column group
      const int k   = kp * 2;
      const int col = c4 * 4;
      const float* gp = &xb[(size_t)(kk * 32 + k) * SP + colbase + col];
      const f32x4 g0 = *(const f32x4*)gp;
      const f32x4 g1 = *(const f32x4*)(gp + SP);
      if (kk + 1 < KT) __builtin_prefetch(gp + 32 * SP, 0, 3);
#pragma unroll
      for (int j = 0; j < 4; ++j) {
        bf16x2 p = { (__bf16)g0[j], (__bf16)g1[j] };
        *(bf16x2*)&sB[buf][(col + j) * 40 + k] = p;
      }
    }
  };

  f32x8 acc[8];
#pragma unroll
  for (int t = 0; t < 8; ++t)
#pragma unroll
    for (int j = 0; j < 8; ++j) acc[t][j] = 0.0f;

  stage(0, 0);
  __syncthreads();

  for (int kk = 0; kk < KT; ++kk) {
    const int cur = kk & 1;
    if (kk + 1 < KT) stage(kk + 1, cur ^ 1);   // overlap with WMMA below

    // A fragment (weights) directly from global fp32, converted, contiguous K (k = 16h + s)
    const float* wr = W + (size_t)arow * C + kk * 32 + h * 16;
    const f32x4 f0 = *(const f32x4*)(wr);
    const f32x4 f1 = *(const f32x4*)(wr + 4);
    const f32x4 f2 = *(const f32x4*)(wr + 8);
    const f32x4 f3 = *(const f32x4*)(wr + 12);
    bf16x16 afrag;
#pragma unroll
    for (int j = 0; j < 4; ++j) {
      afrag[j]      = (__bf16)f0[j];
      afrag[4 + j]  = (__bf16)f1[j];
      afrag[8 + j]  = (__bf16)f2[j];
      afrag[12 + j] = (__bf16)f3[j];
    }

    // preload ALL B fragments, then back-to-back WMMAs
    bf16x16 bfr[8];
#pragma unroll
    for (int t = 0; t < 8; ++t)
      bfr[t] = *(const bf16x16*)&sB[cur][(t * 16 + l16) * 40 + h * 16];
#pragma unroll
    for (int t = 0; t < 8; ++t)
      acc[t] = WMMA_BF16(afrag, bfr[t], acc[t]);

    __syncthreads();
  }

  // epilogue: + bias, store bf16 [channel][spatial]
  float bv8[8];
#pragma unroll
  for (int i = 0; i < 8; ++i) bv8[i] = bias[lrow0 + i + 8 * h];
#pragma unroll
  for (int t = 0; t < 8; ++t) {
    const int col = colbase + t * 16 + l16;
#pragma unroll
    for (int i = 0; i < 8; ++i) {
      const int o = lrow0 + i + 8 * h;
      Out[(size_t)(b * CR + o) * SP + col] = (__bf16)(acc[t][i] + bv8[i]);
    }
  }
}

// ---------------------------------------------------------------------------
// Kernel 2: per-group row attention.  One block (4 waves) per group g=(b,cr).
// S = Q K^T (64x64), row softmax in f32, Y = softmax(S) V.
// ---------------------------------------------------------------------------
__global__ __launch_bounds__(128) void attn_kernel(
    const __bf16* __restrict__ Qo, const __bf16* __restrict__ Ko,
    const __bf16* __restrict__ Vo, __bf16* __restrict__ Yo)
{
  __shared__ __align__(16) __bf16 sQ [64 * 72];  // A-operand, row-major, unswizzled
  __shared__ __align__(16) __bf16 sK [64 * 72];  // B-operand for QK^T (row j = col j)
  __shared__ __align__(16) __bf16 sVt[64 * 72];  // V transposed: B-operand for AV
  __shared__ __align__(16) float  sS [64 * 68];  // scores f32
  __shared__ __align__(16) __bf16 sA [64 * 72];  // softmax(S) bf16, row-major, unswizzled

  const int tid  = threadIdx.x;
  const int lane = tid & 31;
  const int wave = tid >> 5;            // 4 waves
  const size_t base = (size_t)blockIdx.x * SP;
  const __bf16* qg = Qo + base;
  const __bf16* kg = Ko + base;
  const __bf16* vg = Vo + base;

  // stage Q and K (both direct) with async 16B global->LDS copies
#pragma unroll
  for (int i = 0; i < 4; ++i) {
    const int cidx = tid + i * 128;          // 0..511
    const int row  = cidx >> 3;
    const int gi   = cidx & 7;
    ASYNC_CP_B128(qg + row * 64 + gi * 8,
                  &sQ[row * 72 + gi * 8]);
    ASYNC_CP_B128(kg + row * 64 + gi * 8,
                  &sK[row * 72 + gi * 8]);
  }
  // stage V transposed: sVt[n][j] = V[j][n] (element-granular, VGPR path)
#pragma unroll
  for (int i = 0; i < 32; ++i) {
    const int idx = tid + i * 128;           // 0..4095
    const int j = idx >> 6, n = idx & 63;
    sVt[n * 72 + j] = vg[j * 64 + n];
  }
  ASYNC_WAIT();
  __syncthreads();

  const int h = lane >> 4, l16 = lane & 15;
  const int row0 = wave * 16;

  // S = Q K^T
  f32x8 acc[4];
#pragma unroll
  for (int t = 0; t < 4; ++t)
#pragma unroll
    for (int j = 0; j < 8; ++j) acc[t][j] = 0.0f;
#pragma unroll
  for (int kk = 0; kk < 2; ++kk) {
    const bf16x16 afrag = *(const bf16x16*)&sQ[(row0 + l16) * 72 + kk * 32 + h * 16];
    bf16x16 bfr[4];
#pragma unroll
    for (int t = 0; t < 4; ++t)
      bfr[t] = *(const bf16x16*)&sK[(t * 16 + l16) * 72 + kk * 32 + h * 16];
#pragma unroll
    for (int t = 0; t < 4; ++t)
      acc[t] = WMMA_BF16(afrag, bfr[t], acc[t]);
  }
#pragma unroll
  for (int t = 0; t < 4; ++t)
#pragma unroll
    for (int i = 0; i < 8; ++i)
      sS[(row0 + i + 8 * h) * 68 + t * 16 + l16] = acc[t][i];
  __syncthreads();

  // row softmax (f32), write bf16 attn row-major (unswizzled)
  if (tid < 64) {
    const int r = tid;
    float mx = -3.4e38f;
    for (int j = 0; j < 64; ++j) mx = fmaxf(mx, sS[r * 68 + j]);
    float sum = 0.0f;
    for (int j = 0; j < 64; ++j) {
      const float e = __expf(sS[r * 68 + j] - mx);
      sS[r * 68 + j] = e;
      sum += e;
    }
    const float inv = 1.0f / sum;
    for (int j = 0; j < 64; ++j) {
      const int pos = j;
      sA[r * 72 + pos] = (__bf16)(sS[r * 68 + j] * inv);
    }
  }
  __syncthreads();

  // Y = attn @ V
  f32x8 yac[4];
#pragma unroll
  for (int t = 0; t < 4; ++t)
#pragma unroll
    for (int j = 0; j < 8; ++j) yac[t][j] = 0.0f;
#pragma unroll
  for (int kk = 0; kk < 2; ++kk) {
    const bf16x16 afrag = *(const bf16x16*)&sA[(row0 + l16) * 72 + kk * 32 + h * 16];
    bf16x16 bfr[4];
#pragma unroll
    for (int t = 0; t < 4; ++t)
      bfr[t] = *(const bf16x16*)&sVt[(t * 16 + l16) * 72 + kk * 32 + h * 16];
#pragma unroll
    for (int t = 0; t < 4; ++t)
      yac[t] = WMMA_BF16(afrag, bfr[t], yac[t]);
  }
  __bf16* yg = Yo + base;
#pragma unroll
  for (int t = 0; t < 4; ++t)
#pragma unroll
    for (int i = 0; i < 8; ++i)
      yg[(row0 + i + 8 * h) * 64 + t * 16 + l16] = (__bf16)yac[t][i];
}

// ---------------------------------------------------------------------------
// Kernel 3: reconstruct 1x1 conv + BN(eval) + residual.
// out[c,s] = (Wr@Y + br)[c,s]*inv[c] + (beta - mean*inv)[c] + x[c,s]
// ---------------------------------------------------------------------------
__global__ __launch_bounds__(256) void recon_kernel(
    const __bf16* __restrict__ Yo,
    const float* __restrict__ Wr, const float* __restrict__ br,
    const float* __restrict__ gamma, const float* __restrict__ beta,
    const float* __restrict__ rmean, const float* __restrict__ rvar,
    const float* __restrict__ x, float* __restrict__ out)
{
  __shared__ __align__(16) __bf16 sB[2][128 * 40];

  const int tid  = threadIdx.x;
  const int lane = tid & 31;
  const int wave = tid >> 5;
  const int b       = blockIdx.z;
  const int colbase = blockIdx.x * 128;
  const int rowblk  = blockIdx.y * 128;
  const int row0    = rowblk + wave * 16;              // channel base 0..496
  const __bf16* yb  = Yo + (size_t)b * CR * SP;

  const int h = lane >> 4, l16 = lane & 15;
  const int arow = row0 + l16;

  // prefetch the residual x tile this block reads in its epilogue (pull to L2)
#pragma unroll
  for (int i = 0; i < 2; ++i) {
    const int idx = tid + i * 256;     // 512 cachelines of 128B cover 128x128 f32
    const int r = idx >> 2, ls = idx & 3;
    __builtin_prefetch(&x[(size_t)(b * C + rowblk + r) * SP + colbase + ls * 32], 0, 2);
  }

  constexpr int KT = CR / 32;

  // stage Y tile [32 K x 128 cols] bf16 -> [col][K]; bf16x8 loads from two
  // adjacent K-rows, packed b32 stores.
  auto stage = [&](int kk, int buf) {
    const int kp  = tid >> 4;          // 0..15
    const int c8  = tid & 15;          // 8-col group
    const int k   = kp * 2;
    const int col = c8 * 8;
    const __bf16* gp = &yb[(size_t)(kk * 32 + k) * SP + colbase + col];
    const bf16x8 a0 = *(const bf16x8*)gp;
    const bf16x8 a1 = *(const bf16x8*)(gp + SP);
    if (kk + 1 < KT) __builtin_prefetch(gp + 32 * SP, 0, 3);
#pragma unroll
    for (int j = 0; j < 8; ++j) {
      bf16x2 p = { a0[j], a1[j] };
      *(bf16x2*)&sB[buf][(col + j) * 40 + k] = p;
    }
  };

  f32x8 acc[8];
#pragma unroll
  for (int t = 0; t < 8; ++t)
#pragma unroll
    for (int j = 0; j < 8; ++j) acc[t][j] = 0.0f;

  stage(0, 0);
  __syncthreads();

  for (int kk = 0; kk < KT; ++kk) {
    const int cur = kk & 1;
    if (kk + 1 < KT) stage(kk + 1, cur ^ 1);

    const float* wr = Wr + (size_t)arow * CR + kk * 32 + h * 16;
    const f32x4 f0 = *(const f32x4*)(wr);
    const f32x4 f1 = *(const f32x4*)(wr + 4);
    const f32x4 f2 = *(const f32x4*)(wr + 8);
    const f32x4 f3 = *(const f32x4*)(wr + 12);
    bf16x16 afrag;
#pragma unroll
    for (int j = 0; j < 4; ++j) {
      afrag[j]      = (__bf16)f0[j];
      afrag[4 + j]  = (__bf16)f1[j];
      afrag[8 + j]  = (__bf16)f2[j];
      afrag[12 + j] = (__bf16)f3[j];
    }
    bf16x16 bfr[8];
#pragma unroll
    for (int t = 0; t < 8; ++t)
      bfr[t] = *(const bf16x16*)&sB[cur][(t * 16 + l16) * 40 + h * 16];
#pragma unroll
    for (int t = 0; t < 8; ++t)
      acc[t] = WMMA_BF16(afrag, bfr[t], acc[t]);

    __syncthreads();
  }

  // epilogue: fold conv bias + BN affine (hoisted per-channel), add residual
  float invv[8], cstv[8];
#pragma unroll
  for (int i = 0; i < 8; ++i) {
    const int c = row0 + i + 8 * h;
    const float inv = gamma[c] * rsqrtf(rvar[c] + 1e-5f);
    invv[i] = inv;
    cstv[i] = br[c] * inv + beta[c] - rmean[c] * inv;
  }
#pragma unroll
  for (int t = 0; t < 8; ++t) {
    const int col = colbase + t * 16 + l16;
#pragma unroll
    for (int i = 0; i < 8; ++i) {
      const int c = row0 + i + 8 * h;
      const size_t oidx = (size_t)(b * C + c) * SP + col;
      out[oidx] = acc[t][i] * invv[i] + cstv[i] + x[oidx];
    }
  }
}

// ---------------------------------------------------------------------------
extern "C" void kernel_launch(void* const* d_in, const int* in_sizes, int n_in,
                              void* d_out, int out_size, void* d_ws, size_t ws_size,
                              hipStream_t stream) {
  const float* x     = (const float*)d_in[0];
  const float* Wq    = (const float*)d_in[1];
  const float* bq    = (const float*)d_in[2];
  const float* Wk    = (const float*)d_in[3];
  const float* bk    = (const float*)d_in[4];
  const float* Wv    = (const float*)d_in[5];
  const float* bv    = (const float*)d_in[6];
  const float* Wr    = (const float*)d_in[7];
  const float* br    = (const float*)d_in[8];
  const float* gamma = (const float*)d_in[9];
  const float* beta  = (const float*)d_in[10];
  const float* rmean = (const float*)d_in[11];
  const float* rvar  = (const float*)d_in[12];
  float* out = (float*)d_out;

  // workspace: Q,K,V,Y in bf16, each 16*256*4096 elements (33.5 MB) => 134 MB
  const size_t QSZ = (size_t)BATCH * CR * SP;
  __bf16* Qo = (__bf16*)d_ws;
  __bf16* Ko = Qo + QSZ;
  __bf16* Vo = Ko + QSZ;
  __bf16* Yo = Vo + QSZ;

  dim3 g1(SP / 128, (3 * CR) / 128, BATCH);   // (32, 6, 16)
  qkv_proj_kernel<<<g1, 256, 0, stream>>>(x, Wq, bq, Wk, bk, Wv, bv, Qo, Ko, Vo);

  attn_kernel<<<BATCH * CR, 128, 0, stream>>>(Qo, Ko, Vo, Yo);

  dim3 g3(SP / 128, C / 128, BATCH);          // (32, 4, 16)
  recon_kernel<<<g3, 256, 0, stream>>>(Yo, Wr, br, gamma, beta, rmean, rvar, x, out);
}
